// Selection_3856880631980
// MI455X (gfx1250) — matrix operand order repaired and checked by reading, not compiled, not run on hardware
//
#include <hip/hip_runtime.h>
#include <hip/hip_bf16.h>

typedef __attribute__((ext_vector_type(16))) _Float16 v16h;
typedef __attribute__((ext_vector_type(8)))  float    v8f;

#define TWO_LOG2E 2.885390081777927f   // 2*log2(e)

// Prefer the gfx1250 hardware tanh (single TRANS op v_tanh_f32);
// fall back to exp2+rcp identity if the builtin is unavailable.
__device__ __forceinline__ float fast_tanh(float x) {
#if __has_builtin(__builtin_amdgcn_tanhf)
    return __builtin_amdgcn_tanhf(x);
#else
    float p = __builtin_amdgcn_exp2f(x * TWO_LOG2E);
    return 1.0f - 2.0f * __builtin_amdgcn_rcpf(p + 1.0f);
#endif
}

__global__ __launch_bounds__(256) void bool_classifiers_wmma(
    const float* __restrict__ x,   // [B,64]
    const float* __restrict__ W1,  // [8,32,64]
    const float* __restrict__ b1,  // [8,32]
    const float* __restrict__ W2,  // [8,32]
    const float* __restrict__ b2,  // [8]
    float* __restrict__ out,       // [B,8]
    int nTiles)                    // B/16
{
    // W1 staged as A-fragments (hidden x input). frag f = k*4 + t*2 + c,
    // lane slot = f*512 + L*16 halfs. A layout: M = L&15,
    // element j -> K = (L>>4)*8 + (j<8 ? j : j+8).
    __shared__ __align__(32) _Float16 ldsW[32 * 512];   // 32 KB
    // Coefficients per k: [0..31] = b1[k][:], [32..63] = W2[k][:]
    __shared__ __align__(16) float ldsC[8 * 64];        // 2 KB

    const int tid  = threadIdx.x;
    const int lane = tid & 31;
    const int lm   = lane & 15;   // batch row within tile (B/C/D col)
    const int h16  = lane >> 4;

    // ---- stage W1 into LDS as f16 A-fragments (once per block) ----
    #pragma unroll
    for (int s = 0; s < 4; ++s) {
        int slot = tid + s * 256;           // 0..1023
        int f    = slot >> 5;               // fragment 0..31
        int sl   = slot & 31;               // lane within fragment
        int k    = f >> 2;
        int t    = (f >> 1) & 1;            // hidden tile
        int c    = f & 1;                   // input chunk
        int slm  = sl & 15;                 // hidden row M
        int sh   = sl >> 4;                 // K-half
        const float* p = W1 + k * 2048 + (t * 16 + slm) * 64 + c * 32 + sh * 8;
        float4 f0 = *(const float4*)(p);
        float4 f1 = *(const float4*)(p + 4);
        float4 f2 = *(const float4*)(p + 16);
        float4 f3 = *(const float4*)(p + 20);
        _Float16* dst = ldsW + f * 512 + sl * 16;
        dst[0]  = (_Float16)f0.x; dst[1]  = (_Float16)f0.y;
        dst[2]  = (_Float16)f0.z; dst[3]  = (_Float16)f0.w;
        dst[4]  = (_Float16)f1.x; dst[5]  = (_Float16)f1.y;
        dst[6]  = (_Float16)f1.z; dst[7]  = (_Float16)f1.w;
        dst[8]  = (_Float16)f2.x; dst[9]  = (_Float16)f2.y;
        dst[10] = (_Float16)f2.z; dst[11] = (_Float16)f2.w;
        dst[12] = (_Float16)f3.x; dst[13] = (_Float16)f3.y;
        dst[14] = (_Float16)f3.z; dst[15] = (_Float16)f3.w;
    }
    // ---- stage b1 / W2 coefficients ----
    #pragma unroll
    for (int s = 0; s < 2; ++s) {
        int idx = tid + s * 256;            // 0..511
        int k = idx >> 6;
        int j = idx & 63;
        ldsC[idx] = (j < 32) ? b1[k * 32 + j] : W2[k * 32 + (j - 32)];
    }
    __syncthreads();

    const int waveId = (blockIdx.x * blockDim.x + tid) >> 5;
    const int stride = (gridDim.x * blockDim.x) >> 5;

    for (int tile = waveId; tile < nTiles; tile += stride) {
        const int row_base = tile << 4;
        const float* xr = x + (size_t)(row_base + lm) * 64;

        // Opaque zero: defeats LICM so the per-k LDS loads stay inside this
        // loop (one classifier's working set live at a time) instead of being
        // hoisted into 256+ VGPRs and spilled to scratch.
        int zro = 0;
        asm volatile("" : "+v"(zro));
        const _Float16* wbase = ldsW + zro;
        const float*    cbase = ldsC + zro;

        if (tile + stride < nTiles) {
            const float* nx = x + (size_t)(((tile + stride) << 4) + lm) * 64;
            __builtin_prefetch(nx, 0, 1);   // global_prefetch_b8
        }

        // x as B-fragments (input x batch): K = h*8 + (j<8 ? j : j+8) per lane/chunk
        v16h xb[2];
        #pragma unroll
        for (int c = 0; c < 2; ++c) {
            const float* p = xr + c * 32 + h16 * 8;
            float4 f0 = *(const float4*)(p);
            float4 f1 = *(const float4*)(p + 4);
            float4 f2 = *(const float4*)(p + 16);
            float4 f3 = *(const float4*)(p + 20);
            v16h b;
            b[0]  = (_Float16)f0.x; b[1]  = (_Float16)f0.y;
            b[2]  = (_Float16)f0.z; b[3]  = (_Float16)f0.w;
            b[4]  = (_Float16)f1.x; b[5]  = (_Float16)f1.y;
            b[6]  = (_Float16)f1.z; b[7]  = (_Float16)f1.w;
            b[8]  = (_Float16)f2.x; b[9]  = (_Float16)f2.y;
            b[10] = (_Float16)f2.z; b[11] = (_Float16)f2.w;
            b[12] = (_Float16)f3.x; b[13] = (_Float16)f3.y;
            b[14] = (_Float16)f3.z; b[15] = (_Float16)f3.w;
            xb[c] = b;
        }

        float o[8];
        #pragma unroll
        for (int k = 0; k < 8; ++k) {
            const _Float16* wa = wbase + (k * 4) * 512 + lane * 16;
            v16h a00 = *(const v16h*)(wa + 0 * 512);  // t=0,c=0
            v16h a01 = *(const v16h*)(wa + 1 * 512);  // t=0,c=1
            v16h a10 = *(const v16h*)(wa + 2 * 512);  // t=1,c=0
            v16h a11 = *(const v16h*)(wa + 3 * 512);  // t=1,c=1

            const float* ck = cbase + k * 64;
            // bias -> WMMA C operand: C[r] = b1[k][t*16 + r + 8*h16]
            float4 u0 = *(const float4*)(ck + 8 * h16);
            float4 u1 = *(const float4*)(ck + 8 * h16 + 4);
            float4 u2 = *(const float4*)(ck + 16 + 8 * h16);
            float4 u3 = *(const float4*)(ck + 16 + 8 * h16 + 4);
            v8f c0 = {u0.x, u0.y, u0.z, u0.w, u1.x, u1.y, u1.z, u1.w};
            v8f c1 = {u2.x, u2.y, u2.z, u2.w, u3.x, u3.y, u3.z, u3.w};

            v8f acc0 = __builtin_amdgcn_wmma_f32_16x16x32_f16(
                false, a00, false, xb[0], (short)0, c0, false, false);
            acc0 = __builtin_amdgcn_wmma_f32_16x16x32_f16(
                false, a01, false, xb[1], (short)0, acc0, false, false);
            v8f acc1 = __builtin_amdgcn_wmma_f32_16x16x32_f16(
                false, a10, false, xb[0], (short)0, c1, false, false);
            acc1 = __builtin_amdgcn_wmma_f32_16x16x32_f16(
                false, a11, false, xb[1], (short)0, acc1, false, false);

            // W2 coefficients: W2[k][t*16 + r + 8*h16]
            float4 w0  = *(const float4*)(ck + 32 + 8 * h16);
            float4 w1  = *(const float4*)(ck + 32 + 8 * h16 + 4);
            float4 w2c = *(const float4*)(ck + 48 + 8 * h16);
            float4 w3  = *(const float4*)(ck + 48 + 8 * h16 + 4);
            float w2t0[8] = {w0.x, w0.y, w0.z, w0.w, w1.x, w1.y, w1.z, w1.w};
            float w2t1[8] = {w2c.x, w2c.y, w2c.z, w2c.w, w3.x, w3.y, w3.z, w3.w};

            // in-lane dot over this lane's 16 hidden units (bias already in
            // acc); 4 partial sums to break the serial FMA dependency chain
            float s0 = 0.0f, s1 = 0.0f, s2 = 0.0f, s3 = 0.0f;
            #pragma unroll
            for (int r = 0; r < 8; r += 2) {
                s0 += fast_tanh(acc0[r])     * w2t0[r];
                s1 += fast_tanh(acc0[r + 1]) * w2t0[r + 1];
                s2 += fast_tanh(acc1[r])     * w2t1[r];
                s3 += fast_tanh(acc1[r + 1]) * w2t1[r + 1];
            }
            float s = (s0 + s1) + (s2 + s3);
            // merge the two lane-halves (hidden r vs r+8): single shuffle
            s += __shfl_xor(s, 16, 32);

            float z = s + b2[k];
            // (tanh(z)+1)/2 = 1 - 1/(exp2(2z*log2e)+1)
            float p = __builtin_amdgcn_exp2f(z * TWO_LOG2E);
            o[k] = 1.0f - __builtin_amdgcn_rcpf(p + 1.0f);
        }

        // coalesced store: one [row,8] slab per lane (halves hold duplicates)
        if (h16 == 0) {
            float* orow = out + (size_t)(row_base + lm) * 8;
            float4 v0 = {o[0], o[1], o[2], o[3]};
            float4 v1 = {o[4], o[5], o[6], o[7]};
            *(float4*)(orow)     = v0;
            *(float4*)(orow + 4) = v1;
        }
    }
}

extern "C" void kernel_launch(void* const* d_in, const int* in_sizes, int n_in,
                              void* d_out, int out_size, void* d_ws, size_t ws_size,
                              hipStream_t stream) {
    (void)n_in; (void)out_size; (void)d_ws; (void)ws_size;
    const float* x  = (const float*)d_in[0];
    const float* W1 = (const float*)d_in[1];
    const float* b1 = (const float*)d_in[2];
    const float* W2 = (const float*)d_in[3];
    const float* b2 = (const float*)d_in[4];
    // d_in[5] = bias, unused in training-mode forward
    float* out = (float*)d_out;

    const int B = in_sizes[0] / 64;     // x is [B, 64]
    const int nTiles = B >> 4;          // 16 rows per tile

    const int threads = 256;            // 8 waves per block
    int blocks = 512;                   // 4096 waves, ~8 tiles each at B=524288
    int maxBlocks = (nTiles * 32 + threads - 1) / threads;  // >=1 wave per tile
    if (blocks > maxBlocks && maxBlocks > 0) blocks = maxBlocks;
    if (blocks < 1) blocks = 1;

    bool_classifiers_wmma<<<blocks, threads, 0, stream>>>(x, W1, b1, W2, b2, out, nTiles);
}
